// RaTok_79465484910750
// MI455X (gfx1250) — matrix operand order repaired and checked by reading, not compiled, not run on hardware
//
#include <hip/hip_runtime.h>

typedef __attribute__((ext_vector_type(16))) _Float16 v16h;
typedef __attribute__((ext_vector_type(8)))  float    v8f;

#define B_    8
#define S_    512
#define C_    256
#define N_    16
#define RX_   16
#define TX_   12
#define AZ_   256
#define V_    192      // TX*RX
#define KAUG  384      // 2*V  (re/im stacked along K)
#define KSTEPS 12      // KAUG/32
#define TWO_PI 6.28318530717958647692f

// ---------------- Kernel 0: mean over (s,n) per (b,rx) ----------------
__global__ __launch_bounds__(256) void k_mean(const float* __restrict__ adc,
                                              const int* __restrict__ gidx,
                                              float* __restrict__ meanRe,
                                              float* __restrict__ meanIm) {
    int pair = blockIdx.x;             // 0..127 : b*16+rx
    int b = pair >> 4, rx = pair & 15;
    int start = gidx[0] * N_;
    float sre = 0.f, sim = 0.f;
    for (int i = threadIdx.x; i < S_ * N_; i += 256) {
        int s = i >> 4, n = i & 15;
        size_t idx = ((((size_t)b * S_ + s) * C_) + start + n) * 32;
        sre += adc[idx + rx];
        sim += adc[idx + 16 + rx];
    }
    __shared__ float red[2][256];
    red[0][threadIdx.x] = sre; red[1][threadIdx.x] = sim;
    __syncthreads();
    for (int off = 128; off > 0; off >>= 1) {
        if (threadIdx.x < (unsigned)off) {
            red[0][threadIdx.x] += red[0][threadIdx.x + off];
            red[1][threadIdx.x] += red[1][threadIdx.x + off];
        }
        __syncthreads();
    }
    if (threadIdx.x == 0) {
        meanRe[pair] = red[0][0] * (1.0f / 8192.0f);
        meanIm[pair] = red[1][0] * (1.0f / 8192.0f);
    }
}

// ---------------- Kernel 1: 512-pt range FFT per (b,n,rx) ----------------
__global__ __launch_bounds__(256) void k_rangefft(const float* __restrict__ adc,
                                                  const int* __restrict__ gidx,
                                                  const float* __restrict__ meanRe,
                                                  const float* __restrict__ meanIm,
                                                  float* __restrict__ X1re,
                                                  float* __restrict__ X1im) {
    int id = blockIdx.x;               // ((b*16+n)*16+rx)
    int rx = id & 15, n = (id >> 4) & 15, b = id >> 8;
    int start = gidx[0] * N_;
    float mre = meanRe[b * 16 + rx], mim = meanIm[b * 16 + rx];
    __shared__ float lre[512], lim[512];
    int t = threadIdx.x;
    for (int s = t; s < 512; s += 256) {
        size_t idx = (((size_t)b * S_ + s) * C_ + start + n) * 32;
        float w = 0.54f - 0.46f * __cosf(TWO_PI * (float)s / 511.0f);
        int r = __brev((unsigned)s) >> 23;       // 9-bit reversal
        lre[r] = (adc[idx + rx]      - mre) * w;
        lim[r] = (adc[idx + 16 + rx] - mim) * w;
    }
    __syncthreads();
    for (int len = 2; len <= 512; len <<= 1) {
        int half = len >> 1;
        int grp = t / half, pos = t % half;
        int i0 = grp * len + pos, i1 = i0 + half;
        float ang = -TWO_PI * (float)pos / (float)len;
        float wi, wr; __sincosf(ang, &wi, &wr);
        float xr = lre[i1], xi = lim[i1];
        float tr = wr * xr - wi * xi, ti = wr * xi + wi * xr;
        float ur = lre[i0], ui = lim[i0];
        lre[i0] = ur + tr; lim[i0] = ui + ti;
        lre[i1] = ur - tr; lim[i1] = ui - ti;
        __syncthreads();
    }
    // X1 layout: [(b, s, rx, n)] so the Doppler kernel reads 16 contiguous floats
    for (int s = t; s < 512; s += 256) {
        size_t o = (((size_t)b * S_ + s) * RX_ + rx) * N_ + n;
        X1re[o] = lre[s]; X1im[o] = lim[s];
    }
}

// Doppler bin shuffle / A-fragment swizzle helpers ---------------------------
// A stored as [mtile][kstep][lane][16 halves] in the exact CDNA5 16-bit A layout:
//   lane = m + 16*hi, elem e<8 -> kk = hi*8+e ; e>=8 -> kk = 16+hi*8+(e-8)
__device__ __forceinline__ size_t a_slot(int mtile, int kcol, int m) {
    int kstep = kcol >> 5;
    int kk    = kcol & 31;
    int hi    = (kk >> 3) & 1;
    int elem  = (kk & 7) + ((kk >> 4) << 3);
    return (((size_t)mtile * KSTEPS + kstep) * 32 + (m + (hi << 4))) * 16 + elem;
}

// ------- Kernel 2: Hamming(16) + FFT16 + DOPPLER_IDX shuffle -> tiled f16 A -------
__global__ __launch_bounds__(256) void k_doppler(const float* __restrict__ X1re,
                                                 const float* __restrict__ X1im,
                                                 const float* __restrict__ vw,
                                                 _Float16* __restrict__ A) {
    int id = blockIdx.x * 256 + threadIdx.x;   // (b*512+s)*16 + rx
    int rx = id & 15;
    int bs = id >> 4;                          // == mtile (rows bs*16 .. bs*16+15)
    float re[16], im[16];
    const float* pr = X1re + (size_t)id * 16;
    const float* pi = X1im + (size_t)id * 16;
#pragma unroll
    for (int n = 0; n < 16; n++) {
        float w = 0.54f - 0.46f * __cosf(TWO_PI * (float)n / 15.0f);
        re[n] = pr[n] * w; im[n] = pi[n] * w;
    }
    // in-register 16-pt DIT FFT
#pragma unroll
    for (int i = 0; i < 16; i++) {
        int r = ((i & 1) << 3) | ((i & 2) << 1) | ((i & 4) >> 1) | ((i & 8) >> 3);
        if (r > i) {
            float tr = re[i]; re[i] = re[r]; re[r] = tr;
            float ti = im[i]; im[i] = im[r]; im[r] = ti;
        }
    }
#pragma unroll
    for (int len = 2; len <= 16; len <<= 1) {
        int half = len >> 1;
#pragma unroll
        for (int i0 = 0; i0 < 16; i0 += len)
#pragma unroll
            for (int p = 0; p < 8; p++) {
                if (p < half) {
                    float ang = -TWO_PI * (float)p / (float)len;
                    float wi, wr; __sincosf(ang, &wi, &wr);
                    int a = i0 + p, c = a + half;
                    float xr = re[c], xi = im[c];
                    float tr = wr * xr - wi * xi, ti = wr * xi + wi * xr;
                    float ur = re[a], ui = im[a];
                    re[a] = ur + tr; im[a] = ui + ti;
                    re[c] = ur - tr; im[c] = ui - ti;
                }
            }
    }
    const int keep[12] = {0, 5, 6, 7, 8, 9, 10, 11, 12, 13, 14, 15};
#pragma unroll
    for (int t = 0; t < 12; t++) {
        float w = vw[t * 16 + rx];
        int v = t * 16 + rx;
#pragma unroll
        for (int k = 0; k < 16; k++) {
            int ns = (k + keep[t]) & 15;
            A[a_slot(bs, v,      k)] = (_Float16)(re[ns] * w);
            A[a_slot(bs, V_ + v, k)] = (_Float16)(im[ns] * w);
        }
    }
}

// ------- Kernel 3: augmented calib -> f16 B fragments [kstep][atile][lane][16] -------
// lane l holds col a = atile*16 + (l&15), kk = (l>>4)*8 + (elem&7) + (elem>>3)*16
__global__ __launch_bounds__(256) void k_baug(const float* __restrict__ cr,
                                              const float* __restrict__ ci,
                                              _Float16* __restrict__ BRe,
                                              _Float16* __restrict__ BIm) {
    int id = blockIdx.x * 256 + threadIdx.x;   // [0, 2*98304)
    int mtx  = id >= 98304;
    int r    = mtx ? id - 98304 : id;
    int elem = r & 15;
    int lane = (r >> 4) & 31;
    int atile = (r >> 9) & 15;
    int kstep = r >> 13;
    int kk = (elem & 7) + ((lane >> 4) << 3) + ((elem >> 3) << 4);
    int k  = kstep * 32 + kk;
    int a  = atile * 16 + (lane & 15);
    float val;
    if (mtx == 0) val = (k < V_) ? cr[a * V_ + k] : -ci[a * V_ + (k - V_)];
    else          val = (k < V_) ? ci[a * V_ + k] :  cr[a * V_ + (k - V_)];
    (mtx ? BIm : BRe)[r] = (_Float16)val;
}

// ------- Kernel 4: WMMA beamforming GEMM + |.| + transposed store -------
// No LDS, no barriers. Each wave: 2 M-tiles x 4 azimuth tiles; every B fragment
// pair (4x b128) feeds 4 WMMAs, amortizing the load wait 4x.
// grid.x = 256 (8 waves x 2 mtiles), grid.y = 4 (4 azimuth tiles per wave)
__global__ __launch_bounds__(256) void k_gemm(const _Float16* __restrict__ A,
                                              const _Float16* __restrict__ BRe,
                                              const _Float16* __restrict__ BIm,
                                              float* __restrict__ out) {
    int tid  = threadIdx.x;
    int wave = tid >> 5, lane = tid & 31;
    int mt0  = (blockIdx.x * 8 + wave) * 2;
    int at0  = blockIdx.y * 4;
    int hi  = lane >> 4;
    int l16 = lane & 15;
    v8f accRe[2][4] = {}; v8f accIm[2][4] = {};
    const _Float16* aBase0 = A + (size_t)mt0 * KSTEPS * 512;
    const _Float16* aBase1 = aBase0 + (size_t)KSTEPS * 512;

#pragma unroll 2
    for (int kstep = 0; kstep < KSTEPS; kstep++) {
        union { float4 f4[2]; v16h h; } ua0, ua1;
        const float4* pa0 = (const float4*)(aBase0 + ((size_t)kstep * 32 + lane) * 16);
        const float4* pa1 = (const float4*)(aBase1 + ((size_t)kstep * 32 + lane) * 16);
        ua0.f4[0] = pa0[0]; ua0.f4[1] = pa0[1];
        ua1.f4[0] = pa1[0]; ua1.f4[1] = pa1[1];
#pragma unroll
        for (int nt = 0; nt < 4; nt++) {
            size_t boff = (((size_t)kstep * 16 + (at0 + nt)) * 32 + lane) * 16;
            union { float4 f4[2]; v16h h; } ubr, ubi;
            const float4* pbr = (const float4*)(BRe + boff);
            const float4* pbi = (const float4*)(BIm + boff);
            ubr.f4[0] = pbr[0]; ubr.f4[1] = pbr[1];
            ubi.f4[0] = pbi[0]; ubi.f4[1] = pbi[1];
            accRe[0][nt] = __builtin_amdgcn_wmma_f32_16x16x32_f16(
                false, ua0.h, false, ubr.h, (short)0, accRe[0][nt], false, false);
            accRe[1][nt] = __builtin_amdgcn_wmma_f32_16x16x32_f16(
                false, ua1.h, false, ubr.h, (short)0, accRe[1][nt], false, false);
            accIm[0][nt] = __builtin_amdgcn_wmma_f32_16x16x32_f16(
                false, ua0.h, false, ubi.h, (short)0, accIm[0][nt], false, false);
            accIm[1][nt] = __builtin_amdgcn_wmma_f32_16x16x32_f16(
                false, ua1.h, false, ubi.h, (short)0, accIm[1][nt], false, false);
        }
    }
    // magnitude + store: out[b][nn][s][a], lanes contiguous along a; NT stores
#pragma unroll
    for (int m = 0; m < 2; m++) {
#pragma unroll
        for (int nt = 0; nt < 4; nt++) {
#pragma unroll
            for (int r = 0; r < 8; r++) {
                int mrow = (mt0 + m) * 16 + r + (hi << 3);   // global n'
                int nn = mrow & 15;
                int s  = (mrow >> 4) & 511;
                int b  = mrow >> 13;
                int a  = (at0 + nt) * 16 + l16;
                float re = accRe[m][nt][r], im = accIm[m][nt][r];
                __builtin_nontemporal_store(
                    __builtin_sqrtf(re * re + im * im),
                    &out[(((size_t)b * 16 + nn) * 512 + s) * 256 + a]);
            }
        }
    }
}

// ---------------------------------------------------------------------------
extern "C" void kernel_launch(void* const* d_in, const int* in_sizes, int n_in,
                              void* d_out, int out_size, void* d_ws, size_t ws_size,
                              hipStream_t stream) {
    const float* adc = (const float*)d_in[0];
    const float* cr  = (const float*)d_in[1];
    const float* ci  = (const float*)d_in[2];
    const float* vw  = (const float*)d_in[3];
    const int*   gi  = (const int*)d_in[4];
    float* out = (float*)d_out;

    char* ws = (char*)d_ws;
    float*    meanRe = (float*)ws;                              // 128 f32
    float*    meanIm = meanRe + 128;                            // 128 f32
    _Float16* BRe    = (_Float16*)(ws + 1024);                  // 98304 f16
    _Float16* BIm    = BRe + 98304;                             // 98304 f16
    float*    X1re   = (float*)(ws + 394240);                   // 1M f32
    float*    X1im   = X1re + (size_t)B_ * S_ * RX_ * N_;       // 1M f32
    _Float16* A      = (_Float16*)(ws + 8782848);               // 4096*12*32*16 f16

    k_mean    <<<128, 256, 0, stream>>>(adc, gi, meanRe, meanIm);
    k_rangefft<<<B_ * N_ * RX_, 256, 0, stream>>>(adc, gi, meanRe, meanIm, X1re, X1im);
    k_doppler <<<256, 256, 0, stream>>>(X1re, X1im, vw, A);
    k_baug    <<<(2 * 98304) / 256, 256, 0, stream>>>(cr, ci, BRe, BIm);
    k_gemm    <<<dim3(256, 4), 256, 0, stream>>>(A, BRe, BIm, out);
}
